// dgl_evennet_18047452578205
// MI455X (gfx1250) — compile-verified, matrix operand order checked by reading, not run on hardware
//
#include <hip/hip_runtime.h>
#include <hip/hip_bf16.h>

#define HID      128
#define FIN      256
#define NCLS     40
#define NCLS_PAD 48
#define K_EVEN   5

typedef __attribute__((ext_vector_type(16))) _Float16 v16h;
typedef __attribute__((ext_vector_type(8)))  float    v8f;

// ---------------- utility kernels ----------------

__global__ void k_zero(float* __restrict__ p, int n) {
    int i = blockIdx.x * blockDim.x + threadIdx.x;
    if (i < n) p[i] = 0.0f;
}

__global__ void k_zero4(float* __restrict__ p, int n4) {
    int i = blockIdx.x * blockDim.x + threadIdx.x;
    if (i < n4) ((float4*)p)[i] = make_float4(0.f, 0.f, 0.f, 0.f);
}

// in-degree: deg[dst[e]] += 1
__global__ void k_deg(const int* __restrict__ dst, float* __restrict__ deg, int E) {
    int e = blockIdx.x * blockDim.x + threadIdx.x;
    if (e < E) atomicAdd(&deg[dst[e]], 1.0f);
}

// deg -> clamp(deg,1)^-0.5 in place
__global__ void k_isqrt(float* __restrict__ deg, int N) {
    int i = blockIdx.x * blockDim.x + threadIdx.x;
    if (i < N) {
        float d = deg[i];
        deg[i] = rsqrtf(d < 1.0f ? 1.0f : d);
    }
}

// vals[e] = dis[src[e]] * dis[dst[e]]
__global__ void k_vals(const int* __restrict__ src, const int* __restrict__ dst,
                       const float* __restrict__ dis, float* __restrict__ vals, int E) {
    int e = blockIdx.x * blockDim.x + threadIdx.x;
    if (e < E) vals[e] = dis[src[e]] * dis[dst[e]];
}

// f32 -> f16 weight conversion (enc_w: HID x FIN, flat)
__global__ void k_cvt_encw(const float* __restrict__ w, _Float16* __restrict__ wh, int n) {
    int i = blockIdx.x * blockDim.x + threadIdx.x;
    if (i < n) wh[i] = (_Float16)w[i];
}

// dec_w: NCLS x HID -> padded NCLS_PAD x HID f16 (rows >= NCLS zeroed)
__global__ void k_cvt_decw(const float* __restrict__ w, _Float16* __restrict__ wh) {
    int i = blockIdx.x * blockDim.x + threadIdx.x;
    if (i < NCLS_PAD * HID) {
        int r = i / HID, c = i % HID;
        wh[i] = (r < NCLS) ? (_Float16)w[r * HID + c] : (_Float16)0.0f;
    }
}

// ---------------- encoder: h0 = features @ enc_w.T + enc_b ----------------
// grid.x = ceil(N/16) row tiles, block = 256 (8 waves; wave w -> output cols [16w,16w+16)).
// A tile (16x256) staged in LDS as f16, shared by all 8 waves.
__global__ void __launch_bounds__(256)
k_encoder(const float* __restrict__ feat, const _Float16* __restrict__ wh,
          const float* __restrict__ enc_b, const float* __restrict__ theta,
          float* __restrict__ cur, float* __restrict__ zbuf, int nrows) {
    __shared__ _Float16 As[16][FIN];   // 8 KB

    const int row0 = blockIdx.x * 16;
    const int t    = threadIdx.x;

    // stage + convert A tile: thread t -> row t>>4, 16 consecutive k
    {
        int r    = t >> 4;
        int kc   = (t & 15) * 16;
        int grow = row0 + r;
        const float* fr = feat + (size_t)grow * FIN;
#pragma unroll
        for (int j = 0; j < 16; ++j) {
            float v = (grow < nrows) ? fr[kc + j] : 0.0f;
            As[r][kc + j] = (_Float16)v;
        }
    }
    __syncthreads();

    const int wave = t >> 5;        // col tile 0..7
    const int lane = t & 31;
    const int q    = lane >> 4;     // half-wave selector (K phase)
    const int ln   = lane & 15;     // M (for A) / N (for B,D)
    const int ncol = wave * 16 + ln;

    const _Float16* brow = wh + (size_t)ncol * FIN;   // B column ncol = enc_w row ncol

    v8f acc = {};
    for (int k0 = 0; k0 < FIN; k0 += 32) {
        v16h a, b;
#pragma unroll
        for (int j = 0; j < 8; ++j) {
            a[j]     = As[ln][k0 + q * 8 + j];        // K = k0 + q*8 + j
            a[j + 8] = As[ln][k0 + 16 + q * 8 + j];   // K = k0 + 16 + q*8 + j
            b[j]     = brow[k0 + q * 8 + j];
            b[j + 8] = brow[k0 + 16 + q * 8 + j];
        }
        acc = __builtin_amdgcn_wmma_f32_16x16x32_f16(
                  false, a, false, b, (short)0, acc, false, false);
    }

    const float th0  = theta[0];
    const float bias = enc_b[ncol];
#pragma unroll
    for (int i = 0; i < 8; ++i) {
        int r = row0 + i + 8 * q;   // D: VGPR i holds M=i (lanes 0-15), M=i+8 (lanes 16-31)
        if (r < nrows) {
            float  h   = acc[i] + bias;
            size_t idx = (size_t)r * HID + ncol;
            cur[idx]  = h;
            zbuf[idx] = th0 * h;    // Z = theta[0] * h0
        }
    }
}

// ---------------- SpMM: out[src[e]] += vals[e] * x[dst[e]] ----------------
// one edge per 32 lanes; float4 per lane (128 features), atomic scatter.
__global__ void __launch_bounds__(256)
k_spmm(const float* __restrict__ vals, const int* __restrict__ src,
       const int* __restrict__ dst, const float* __restrict__ xin,
       float* __restrict__ xout, int E) {
    long long gid = (long long)blockIdx.x * blockDim.x + threadIdx.x;
    int e    = (int)(gid >> 5);
    int lane = (int)(gid & 31);
    if (e >= E) return;
    int   s = src[e];
    int   d = dst[e];
    float v = vals[e];
    float4 x = ((const float4*)(xin + (size_t)d * HID))[lane];
    float* o = xout + (size_t)s * HID + lane * 4;
    atomicAdd(o + 0, v * x.x);
    atomicAdd(o + 1, v * x.y);
    atomicAdd(o + 2, v * x.z);
    atomicAdd(o + 3, v * x.w);
}

// Z += theta[k] * cur   (float4 vectorized)
__global__ void k_axpy(float* __restrict__ z, const float* __restrict__ c,
                       const float* __restrict__ theta, int k, int n4) {
    int i = blockIdx.x * blockDim.x + threadIdx.x;
    if (i >= n4) return;
    float  t  = theta[k];
    float4 zz = ((float4*)z)[i];
    float4 cc = ((const float4*)c)[i];
    zz.x += t * cc.x; zz.y += t * cc.y; zz.z += t * cc.z; zz.w += t * cc.w;
    ((float4*)z)[i] = zz;
}

// ---------------- decoder: logits = Z @ dec_w.T + dec_b ----------------
// grid.x = row tiles, block = 96 (3 waves -> 48 padded output cols), store only col<40.
__global__ void __launch_bounds__(96)
k_decoder(const float* __restrict__ zbuf, const _Float16* __restrict__ wh,
          const float* __restrict__ dec_b, float* __restrict__ out, int nrows) {
    __shared__ _Float16 As[16][HID];   // 4 KB

    const int row0 = blockIdx.x * 16;
    const int t    = threadIdx.x;

    for (int idx = t; idx < 16 * HID; idx += 96) {
        int r = idx / HID, c = idx % HID;
        int grow = row0 + r;
        float v = (grow < nrows) ? zbuf[(size_t)grow * HID + c] : 0.0f;
        As[r][c] = (_Float16)v;
    }
    __syncthreads();

    const int wave = t >> 5;            // 0..2
    const int lane = t & 31;
    const int q    = lane >> 4;
    const int ln   = lane & 15;
    const int ncol = wave * 16 + ln;    // 0..47 (padded)

    const _Float16* brow = wh + (size_t)ncol * HID;

    v8f acc = {};
    for (int k0 = 0; k0 < HID; k0 += 32) {
        v16h a, b;
#pragma unroll
        for (int j = 0; j < 8; ++j) {
            a[j]     = As[ln][k0 + q * 8 + j];
            a[j + 8] = As[ln][k0 + 16 + q * 8 + j];
            b[j]     = brow[k0 + q * 8 + j];
            b[j + 8] = brow[k0 + 16 + q * 8 + j];
        }
        acc = __builtin_amdgcn_wmma_f32_16x16x32_f16(
                  false, a, false, b, (short)0, acc, false, false);
    }

    if (ncol < NCLS) {
        float bias = dec_b[ncol];
#pragma unroll
        for (int i = 0; i < 8; ++i) {
            int r = row0 + i + 8 * q;
            if (r < nrows) out[(size_t)r * NCLS + ncol] = acc[i] + bias;
        }
    }
}

// ---------------- host launch ----------------

extern "C" void kernel_launch(void* const* d_in, const int* in_sizes, int n_in,
                              void* d_out, int out_size, void* d_ws, size_t ws_size,
                              hipStream_t stream) {
    const float* features = (const float*)d_in[0];
    const int*   src      = (const int*)d_in[1];
    const int*   dst      = (const int*)d_in[2];
    const float* enc_w    = (const float*)d_in[3];
    const float* enc_b    = (const float*)d_in[4];
    const float* theta    = (const float*)d_in[5];
    const float* dec_w    = (const float*)d_in[6];
    const float* dec_b    = (const float*)d_in[7];
    float*       out      = (float*)d_out;

    const int N = in_sizes[0] / FIN;    // 100000
    const int E = in_sizes[1];          // 1600000

    // workspace carve-up (~161 MB)
    char* ws = (char*)d_ws;
    auto al = [](size_t x) { return (x + 255) & ~(size_t)255; };
    size_t off = 0;
    float* deg  = (float*)(ws + off); off += al((size_t)N * 4);            // deg -> d_isqrt
    float* vals = (float*)(ws + off); off += al((size_t)E * 4);
    float* cur  = (float*)(ws + off); off += al((size_t)N * HID * 4);
    float* tmp  = (float*)(ws + off); off += al((size_t)N * HID * 4);
    float* zbuf = (float*)(ws + off); off += al((size_t)N * HID * 4);
    _Float16* encw_h = (_Float16*)(ws + off); off += al((size_t)HID * FIN * 2);
    _Float16* decw_h = (_Float16*)(ws + off); off += al((size_t)NCLS_PAD * HID * 2);
    (void)ws_size; (void)n_in; (void)out_size;

    const int B = 256;
    const int nh  = N * HID;
    const int nh4 = nh / 4;

    // 1) edge normalization
    k_zero<<<(N + B - 1) / B, B, 0, stream>>>(deg, N);
    k_deg<<<(E + B - 1) / B, B, 0, stream>>>(dst, deg, E);
    k_isqrt<<<(N + B - 1) / B, B, 0, stream>>>(deg, N);
    k_vals<<<(E + B - 1) / B, B, 0, stream>>>(src, dst, deg, vals, E);

    // 2) weight conversion to f16 (decoder padded to 48 rows)
    k_cvt_encw<<<(HID * FIN + B - 1) / B, B, 0, stream>>>(enc_w, encw_h, HID * FIN);
    k_cvt_decw<<<(NCLS_PAD * HID + B - 1) / B, B, 0, stream>>>(dec_w, decw_h);

    // 3) encoder GEMM (WMMA), writes cur = h0 and zbuf = theta[0]*h0
    const int mtiles = (N + 15) / 16;
    k_encoder<<<mtiles, 256, 0, stream>>>(features, encw_h, enc_b, theta, cur, zbuf, N);

    // 4) 5 x (P @ P @ cur), accumulate into Z
    const int spmm_blocks = (int)(((long long)E * 32 + B - 1) / B);
    for (int k = 1; k <= K_EVEN; ++k) {
        k_zero4<<<(nh4 + B - 1) / B, B, 0, stream>>>(tmp, nh4);
        k_spmm<<<spmm_blocks, B, 0, stream>>>(vals, src, dst, cur, tmp, E);
        k_zero4<<<(nh4 + B - 1) / B, B, 0, stream>>>(cur, nh4);
        k_spmm<<<spmm_blocks, B, 0, stream>>>(vals, src, dst, tmp, cur, E);
        k_axpy<<<(nh4 + B - 1) / B, B, 0, stream>>>(zbuf, cur, theta, k, nh4);
    }

    // 5) decoder GEMM (WMMA) + bias -> logits
    k_decoder<<<mtiles, 96, 0, stream>>>(zbuf, decw_h, dec_b, out, N);
}
